// ConditionalJiTBlock_40862318854218
// MI455X (gfx1250) — matrix operand order repaired and checked by reading, not compiled, not run on hardware
//
#include <hip/hip_runtime.h>

// ---------------------------------------------------------------------------
// Problem constants (from reference)
// ---------------------------------------------------------------------------
#define B_   4
#define N_   1024
#define M_   1024
#define D_   1024
#define H_   16
#define HD_  64
#define MH_  2730
#define MHP_ 2752              // MH padded to multiple of 64 (zero-filled)
#define NINE_D (9 * D_)
#define EPS_ 1e-6f

typedef __attribute__((ext_vector_type(16))) _Float16 v16h;
typedef __attribute__((ext_vector_type(8)))  _Float16 v8h;
typedef __attribute__((ext_vector_type(4)))  _Float16 v4h;
typedef __attribute__((ext_vector_type(8)))  float    v8f;

// ---------------------------------------------------------------------------
// CDNA5 helpers
// ---------------------------------------------------------------------------
__device__ __forceinline__ void wait_dscnt0() {
#if __has_builtin(__builtin_amdgcn_s_wait_dscnt)
  __builtin_amdgcn_s_wait_dscnt(0);
#else
  asm volatile("s_wait_dscnt 0" ::: "memory");
#endif
}

__device__ __forceinline__ v8f wmma_f16(v16h a, v16h b, v8f c) {
  return __builtin_amdgcn_wmma_f32_16x16x32_f16(
      false, a, false, b, (short)0, c, false, false);
}

__device__ __forceinline__ v16h cat8(v8h lo, v8h hi) {
  return __builtin_shufflevector(lo, hi, 0, 1, 2, 3, 4, 5, 6, 7,
                                         8, 9, 10, 11, 12, 13, 14, 15);
}

// A-fragment (16x32 f16, row-major, stride ld; ld % 8 == 0, base 16B aligned).
// A layout (contiguous K): lanes 0-15 row=lane K=0..15; lanes 16-31 K=16..31.
__device__ __forceinline__ v16h frag_load_A16(const _Float16* base, int ld, int lane) {
  int row = lane & 15;
  int kh  = (lane >> 4) * 16;
  const _Float16* r = base + (size_t)row * ld + kh;
  v8h lo = *(const v8h*)r;          // K = kh .. kh+7
  v8h hi = *(const v8h*)(r + 8);    // K = kh+8 .. kh+15
  return cat8(lo, hi);
}

// B-fragment (32x16 f16) from a K-contiguous ("transposed") buffer:
// element (k, col) at baseT[col*ldt + k].  lanes 0-15: col=lane, K=0..15;
// lanes 16-31: col=lane-16, K=16..31.
__device__ __forceinline__ v16h frag_load_BT(const _Float16* baseT, int ldt, int lane) {
  int col = lane & 15;
  int k0  = (lane >> 4) * 16;
  const _Float16* r = baseT + (size_t)col * ldt + k0;
  v8h lo = *(const v8h*)r;
  v8h hi = *(const v8h*)(r + 8);
  return cat8(lo, hi);
}

// ---------------------------------------------------------------------------
// Elementwise / small kernels
// ---------------------------------------------------------------------------
__global__ __launch_bounds__(256)
void cast4_f32_f16(const float4* __restrict__ in, v4h* __restrict__ out, size_t n4) {
  size_t i = (size_t)blockIdx.x * 256 + threadIdx.x;
  size_t stride = (size_t)gridDim.x * 256;
  for (; i < n4; i += stride) {
    float4 v = in[i];
    v4h o = {(_Float16)v.x, (_Float16)v.y, (_Float16)v.z, (_Float16)v.w};
    out[i] = o;
  }
}

// Zero-padding cast: src is (srows x scols) f32, dst is (drows x dcols) f16.
__global__ __launch_bounds__(256)
void cast_pad_f32_f16(const float* __restrict__ src, _Float16* __restrict__ dst,
                      int srows, int scols, int drows, int dcols) {
  size_t total = (size_t)drows * dcols;
  size_t i = (size_t)blockIdx.x * 256 + threadIdx.x;
  size_t stride = (size_t)gridDim.x * 256;
  for (; i < total; i += stride) {
    int r = (int)(i / dcols), c = (int)(i % dcols);
    dst[i] = (r < srows && c < scols) ? (_Float16)src[(size_t)r * scols + c]
                                      : (_Float16)0.f;
  }
}

__global__ __launch_bounds__(256)
void pad_bias_f32(const float* __restrict__ src, float* __restrict__ dst,
                  int n, int np) {
  int i = blockIdx.x * 256 + threadIdx.x;
  if (i < np) dst[i] = (i < n) ? src[i] : 0.f;
}

// mods = silu(c) @ ada_w + ada_b   (tiny: 4 x 9216, K=1024)
__global__ __launch_bounds__(256)
void mods_kernel(const float* __restrict__ c, const float* __restrict__ ada_w,
                 const float* __restrict__ ada_b, float* __restrict__ mods) {
  int j = blockIdx.x * 256 + threadIdx.x;
  int b = blockIdx.y;
  if (j >= NINE_D) return;
  float acc = ada_b[j];
  const float* cr = c + (size_t)b * D_;
  for (int d = 0; d < D_; ++d) {
    float cv = cr[d];
    acc += (cv / (1.f + __expf(-cv))) * ada_w[(size_t)d * NINE_D + j];
  }
  mods[(size_t)b * NINE_D + j] = acc;
}

// xn = modulate(rms(x,w), shift, scale) cast to f16.  One block per token row.
__global__ __launch_bounds__(256)
void rms_mod_cast(const float* __restrict__ x, const float* __restrict__ w,
                  const float* __restrict__ mods, int shift_off, int scale_off,
                  _Float16* __restrict__ out) {
  int row = blockIdx.x;            // b*N + n
  int b   = row / N_;
  const float4* xr = (const float4*)(x + (size_t)row * D_);
  float ss = 0.f;
  for (int i = threadIdx.x; i < D_ / 4; i += 256) {
    float4 v = xr[i];
    ss += v.x * v.x + v.y * v.y + v.z * v.z + v.w * v.w;
  }
#pragma unroll
  for (int o = 16; o > 0; o >>= 1) ss += __shfl_xor(ss, o, 32);
  __shared__ float red[8];
  if ((threadIdx.x & 31) == 0) red[threadIdx.x >> 5] = ss;
  __syncthreads();
  float tot = 0.f;
#pragma unroll
  for (int i = 0; i < 8; ++i) tot += red[i];
  float scl = rsqrtf(tot * (1.f / D_) + EPS_);
  const float4* wr = (const float4*)w;
  const float4* sh = (const float4*)(mods + (size_t)b * NINE_D + shift_off);
  const float4* sc = (const float4*)(mods + (size_t)b * NINE_D + scale_off);
  v4h* orow = (v4h*)(out + (size_t)row * D_);
  for (int i = threadIdx.x; i < D_ / 4; i += 256) {
    float4 v = xr[i], ww = wr[i], s1 = sh[i], s2 = sc[i];
    v4h o = {(_Float16)(v.x * scl * ww.x * (1.f + s2.x) + s1.x),
             (_Float16)(v.y * scl * ww.y * (1.f + s2.y) + s1.y),
             (_Float16)(v.z * scl * ww.z * (1.f + s2.z) + s1.z),
             (_Float16)(v.w * scl * ww.w * (1.f + s2.w) + s1.w)};
    orow[i] = o;
  }
}

// Per-head RMSNorm (or plain cast when w==nullptr).  One wave per (token, head).
// transpose==0: out head-major (B,H,nTok,HD).  transpose==1: out (B,H,HD,nTok).
__global__ __launch_bounds__(256)
void head_rms_cast(const float* __restrict__ src, int tok_stride, int head_off,
                   const float* __restrict__ w, _Float16* __restrict__ out,
                   int nTok, int transpose) {
  int wave = threadIdx.x >> 5, lane = threadIdx.x & 31;
  int gw = blockIdx.x * 8 + wave;                // (b*nTok + n)*H + h
  int h  = gw % H_;
  int t  = gw / H_;                              // b*nTok + n
  int b  = t / nTok, n = t % nTok;
  const float* s = src + (size_t)t * tok_stride + head_off + h * HD_;
  float v0 = s[lane], v1 = s[lane + 32];
  float r = 1.f, w0 = 1.f, w1 = 1.f;
  if (w) {
    float ss = v0 * v0 + v1 * v1;
#pragma unroll
    for (int o = 16; o > 0; o >>= 1) ss += __shfl_xor(ss, o, 32);
    r  = rsqrtf(ss * (1.f / HD_) + EPS_);
    w0 = w[lane]; w1 = w[lane + 32];
  }
  size_t head = (size_t)(b * H_ + h);
  if (transpose) {
    _Float16* o_ = out + (head * HD_) * nTok + n;
    o_[(size_t)lane * nTok]        = (_Float16)(v0 * r * w0);
    o_[(size_t)(lane + 32) * nTok] = (_Float16)(v1 * r * w1);
  } else {
    _Float16* o_ = out + (head * nTok + n) * HD_;
    o_[lane]      = (_Float16)(v0 * r * w0);
    o_[lane + 32] = (_Float16)(v1 * r * w1);
  }
}

// x_out = x_in + gate(b,col) * proj   (float4 vectorized)
__global__ __launch_bounds__(256)
void residual_gate4(const float4* __restrict__ xin, const float4* __restrict__ proj,
                    const float* __restrict__ mods, int gate_off,
                    float4* __restrict__ xout) {
  size_t i = (size_t)blockIdx.x * 256 + threadIdx.x;
  if (i >= (size_t)B_ * N_ * D_ / 4) return;
  int b    = (int)(i >> 18);                    // (N_*D_)/4 = 1<<18
  int col4 = (int)(i & (D_ / 4 - 1));
  float4 g = *(const float4*)(mods + (size_t)b * NINE_D + gate_off + col4 * 4);
  float4 xi = xin[i], pr = proj[i];
  float4 o = {xi.x + g.x * pr.x, xi.y + g.y * pr.y,
              xi.z + g.z * pr.z, xi.w + g.w * pr.w};
  xout[i] = o;
}

// hsw = silu(h1) * h2  (f32 -> f16, float4 vectorized)
__global__ __launch_bounds__(256)
void swiglu4(const float4* __restrict__ h1, const float4* __restrict__ h2,
             v4h* __restrict__ out, size_t n4) {
  size_t i = (size_t)blockIdx.x * 256 + threadIdx.x;
  size_t stride = (size_t)gridDim.x * 256;
  for (; i < n4; i += stride) {
    float4 a = h1[i], b = h2[i];
    v4h o = {(_Float16)(a.x / (1.f + __expf(-a.x)) * b.x),
             (_Float16)(a.y / (1.f + __expf(-a.y)) * b.y),
             (_Float16)(a.z / (1.f + __expf(-a.z)) * b.z),
             (_Float16)(a.w / (1.f + __expf(-a.w)) * b.w)};
    out[i] = o;
  }
}

// ---------------------------------------------------------------------------
// WMMA GEMM: C(f32, MxN) = A(f16, MxK) @ B(f16, KxN) + bias
// Requires M%128==0, N%64==0, K%32==0 (ragged dims are pre-padded with zeros).
// Block: 256 threads = 8 waves, tile 128x64, BK=32.
// Waves 4x2; each wave computes 32x32 via 4 WMMA accumulators.
// ---------------------------------------------------------------------------
#define BM 128
#define BN 64
#define BK 32
#define LDA_S  40    // BK + 8 (multiple of 8 -> rows stay 16B aligned)
#define LDBT_S 40

__global__ __launch_bounds__(256)
void gemm_f16f32(const _Float16* __restrict__ A, const _Float16* __restrict__ Bw,
                 const float* __restrict__ bias, float* __restrict__ C,
                 int M, int N, int K) {
  __shared__ __align__(16) _Float16 As[BM][LDA_S];    // 10 KB
  __shared__ __align__(16) _Float16 BsT[BN][LDBT_S];  //  5 KB, [col][k]
  int tid  = threadIdx.x;
  int lane = tid & 31, wave = tid >> 5;
  int wm = (wave & 3) * 32;
  int wn = (wave >> 2) * 32;
  int bm0 = blockIdx.y * BM;
  int bn0 = blockIdx.x * BN;
  v8f acc00 = {}, acc01 = {}, acc10 = {}, acc11 = {};

  // staging coordinates
  int ar = tid >> 1;             // 0..127 (A row)
  int ac = (tid & 1) * 16;       // 0 / 16 (A col group)
  int br = tid >> 3;             // 0..31  (B k-row)
  int bc = (tid & 7) * 8;        // B col group
  const _Float16* aSrc = A + (size_t)(bm0 + ar) * K + ac;
  const _Float16* bSrc = Bw + (size_t)br * N + bn0 + bc;

  for (int k0 = 0; k0 < K; k0 += BK) {
    // A tile 128x32: two aligned 16B vectors per thread
    v8h av0 = *(const v8h*)(aSrc + k0);
    v8h av1 = *(const v8h*)(aSrc + k0 + 8);
    *(v8h*)&As[ar][ac]     = av0;
    *(v8h*)&As[ar][ac + 8] = av1;
    // B tile 32x64: aligned 16B global load, transposed LDS store
    v8h bv = *(const v8h*)(bSrc + (size_t)k0 * N);
#pragma unroll
    for (int i = 0; i < 8; ++i) BsT[bc + i][br] = bv[i];
    if (k0 + BK < K) {
      __builtin_prefetch(aSrc + k0 + BK, 0, 1);                 // global_prefetch
      __builtin_prefetch(bSrc + (size_t)(k0 + BK) * N, 0, 1);
    }
    __syncthreads();

    v16h af0 = frag_load_A16(&As[wm][0],      LDA_S,  lane);
    v16h af1 = frag_load_A16(&As[wm + 16][0], LDA_S,  lane);
    v16h bf0 = frag_load_BT(&BsT[wn][0],      LDBT_S, lane);
    v16h bf1 = frag_load_BT(&BsT[wn + 16][0], LDBT_S, lane);
    acc00 = wmma_f16(af0, bf0, acc00);
    acc01 = wmma_f16(af0, bf1, acc01);
    acc10 = wmma_f16(af1, bf0, acc10);
    acc11 = wmma_f16(af1, bf1, acc11);
    __syncthreads();
  }

  // epilogue (C layout: lanes 0-15 rows mofs..mofs+7, col = lane&15)
  int nn = lane & 15, mofs = (lane >> 4) * 8;
  int col0 = bn0 + wn + nn, col1 = col0 + 16;
  float bi0 = bias ? bias[col0] : 0.f;
  float bi1 = bias ? bias[col1] : 0.f;
#pragma unroll
  for (int r = 0; r < 8; ++r) {
    int m0 = bm0 + wm + mofs + r;
    int m1 = m0 + 16;
    C[(size_t)m0 * N + col0] = acc00[r] + bi0;
    C[(size_t)m0 * N + col1] = acc01[r] + bi1;
    C[(size_t)m1 * N + col0] = acc10[r] + bi0;
    C[(size_t)m1 * N + col1] = acc11[r] + bi1;
  }
}

// ---------------------------------------------------------------------------
// Flash attention, one wave per (b, h, 16-query tile).  HD=64.
// Q/K head-major f16 (B,H,nTok,64).  V transposed f16 (B,H,64,Nk).
// Out token-major f16 (B,Nq,H*64).
// ---------------------------------------------------------------------------
__global__ __launch_bounds__(32)
void flash_attn(const _Float16* __restrict__ Q, const _Float16* __restrict__ Kh,
                const _Float16* __restrict__ Vt, _Float16* __restrict__ Out,
                int Nq, int Nk, float scale) {
  int lane = threadIdx.x;
  int qt = blockIdx.x, h = blockIdx.y, b = blockIdx.z;
  const _Float16* q  = Q  + ((size_t)(b * H_ + h) * Nq + (size_t)qt * 16) * HD_;
  const _Float16* k  = Kh + (size_t)(b * H_ + h) * Nk * HD_;
  const _Float16* vt = Vt + (size_t)(b * H_ + h) * HD_ * Nk;

  v16h a0 = frag_load_A16(q, HD_, lane);        // Q dims 0..31
  v16h a1 = frag_load_A16(q + 32, HD_, lane);   // Q dims 32..63

  float m_i[8], l_i[8];
  v8f o[4] = {{}, {}, {}, {}};
#pragma unroll
  for (int r = 0; r < 8; ++r) { m_i[r] = -1e30f; l_i[r] = 0.f; }

  __shared__ __align__(16) _Float16 Ps[16][40];
  int nn = lane & 15, mofs = (lane >> 4) * 8;
  int kf = (lane >> 4) * 16;

  for (int j0 = 0; j0 < Nk; j0 += 32) {
    // S = Q @ K^T for a 16x32 key chunk (two 16-col subtiles)
    v8f s[2];
#pragma unroll
    for (int c = 0; c < 2; ++c) {
      const _Float16* kr = k + (size_t)(j0 + c * 16 + nn) * HD_ + kf;
      v16h b0 = cat8(*(const v8h*)kr,        *(const v8h*)(kr + 8));   // dims kf..
      v16h b1 = cat8(*(const v8h*)(kr + 32), *(const v8h*)(kr + 40));  // dims 32+kf..
      v8f sc = {};
      sc = wmma_f16(a0, b0, sc);
      sc = wmma_f16(a1, b1, sc);
      s[c] = sc;
    }
    // Online softmax across the 32 columns (rows live in lane-halves of 16)
#pragma unroll
    for (int r = 0; r < 8; ++r) {
      float v0 = s[0][r] * scale, v1 = s[1][r] * scale;
      float mx = fmaxf(v0, v1);
#pragma unroll
      for (int of = 8; of > 0; of >>= 1) mx = fmaxf(mx, __shfl_xor(mx, of, 16));
      float mn = fmaxf(m_i[r], mx);
      float alpha = __expf(m_i[r] - mn);
      float p0 = __expf(v0 - mn);
      float p1 = __expf(v1 - mn);
      float rs = p0 + p1;
#pragma unroll
      for (int of = 8; of > 0; of >>= 1) rs += __shfl_xor(rs, of, 16);
      l_i[r] = l_i[r] * alpha + rs;
      m_i[r] = mn;
      o[0][r] *= alpha; o[1][r] *= alpha; o[2][r] *= alpha; o[3][r] *= alpha;
      Ps[mofs + r][nn]      = (_Float16)p0;
      Ps[mofs + r][16 + nn] = (_Float16)p1;
    }
    wait_dscnt0();   // P tile visible in LDS before re-reading as A-fragment
    v16h pa = frag_load_A16(&Ps[0][0], 40, lane);
    // O += P(16x32) @ V(32x64): V rows contiguous in Vt -> b128 loads
#pragma unroll
    for (int g = 0; g < 4; ++g) {
      const _Float16* vr = vt + (size_t)(g * 16 + nn) * Nk + j0 + kf;
      v16h bv = cat8(*(const v8h*)vr, *(const v8h*)(vr + 8));
      o[g] = wmma_f16(pa, bv, o[g]);
    }
  }

  // epilogue: normalize and write token-major (B, Nq, H*HD)
  size_t row0 = (size_t)b * Nq + (size_t)qt * 16 + mofs;
#pragma unroll
  for (int r = 0; r < 8; ++r) {
    float inv = 1.f / l_i[r];
    size_t base = (row0 + r) * (size_t)(H_ * HD_) + (size_t)h * HD_;
#pragma unroll
    for (int g = 0; g < 4; ++g)
      Out[base + g * 16 + nn] = (_Float16)(o[g][r] * inv);
  }
}

// ---------------------------------------------------------------------------
// Host driver
// ---------------------------------------------------------------------------
extern "C" void kernel_launch(void* const* d_in, const int* in_sizes, int n_in,
                              void* d_out, int out_size, void* d_ws, size_t ws_size,
                              hipStream_t stream) {
  const float* x        = (const float*)d_in[0];
  const float* src_tok  = (const float*)d_in[1];
  const float* c        = (const float*)d_in[2];
  const float* ada_w    = (const float*)d_in[3];
  const float* ada_b    = (const float*)d_in[4];
  const float* n1_w     = (const float*)d_in[5];
  const float* nc_w     = (const float*)d_in[6];
  const float* n2_w     = (const float*)d_in[7];
  const float* sa_qkv_w = (const float*)d_in[8];
  const float* sa_qkv_b = (const float*)d_in[9];
  const float* sa_qn_w  = (const float*)d_in[10];
  const float* sa_kn_w  = (const float*)d_in[11];
  const float* sa_o_w   = (const float*)d_in[12];
  const float* sa_o_b   = (const float*)d_in[13];
  const float* ca_q_w   = (const float*)d_in[14];
  const float* ca_q_b   = (const float*)d_in[15];
  const float* ca_kv_w  = (const float*)d_in[16];
  const float* ca_kv_b  = (const float*)d_in[17];
  const float* ca_qn_w  = (const float*)d_in[18];
  const float* ca_kn_w  = (const float*)d_in[19];
  const float* ca_o_w   = (const float*)d_in[20];
  const float* ca_o_b   = (const float*)d_in[21];
  const float* mlp_w1   = (const float*)d_in[22];
  const float* mlp_b1   = (const float*)d_in[23];
  const float* mlp_w2   = (const float*)d_in[24];
  const float* mlp_b2   = (const float*)d_in[25];
  const float* mlp_w3   = (const float*)d_in[26];
  const float* mlp_b3   = (const float*)d_in[27];
  float* out = (float*)d_out;

  const int BN_TOK = B_ * N_;                 // 4096 token rows

  char* p = (char*)d_ws;
  auto alloc = [&](size_t bytes) -> char* {
    char* r = p; p += (bytes + 255) & ~(size_t)255; return r;
  };
  // f16 weights (MLP dims padded to MHP_ with zeros)
  _Float16* w_qkv_h  = (_Float16*)alloc(sizeof(_Float16) * (size_t)D_ * 3 * D_);
  _Float16* w_sao_h  = (_Float16*)alloc(sizeof(_Float16) * (size_t)D_ * D_);
  _Float16* w_caq_h  = (_Float16*)alloc(sizeof(_Float16) * (size_t)D_ * D_);
  _Float16* w_cakv_h = (_Float16*)alloc(sizeof(_Float16) * (size_t)D_ * 2 * D_);
  _Float16* w_cao_h  = (_Float16*)alloc(sizeof(_Float16) * (size_t)D_ * D_);
  _Float16* w_m1_h   = (_Float16*)alloc(sizeof(_Float16) * (size_t)D_ * MHP_);
  _Float16* w_m2_h   = (_Float16*)alloc(sizeof(_Float16) * (size_t)D_ * MHP_);
  _Float16* w_m3_h   = (_Float16*)alloc(sizeof(_Float16) * (size_t)MHP_ * D_);
  float*    b1p      = (float*)alloc(sizeof(float) * MHP_);
  float*    b2p      = (float*)alloc(sizeof(float) * MHP_);
  // f32 scratch
  float* mods  = (float*)alloc(sizeof(float) * (size_t)B_ * NINE_D);
  float* x1    = (float*)alloc(sizeof(float) * (size_t)BN_TOK * D_);
  float* x2    = (float*)alloc(sizeof(float) * (size_t)BN_TOK * D_);
  float* qkv   = (float*)alloc(sizeof(float) * (size_t)BN_TOK * 3 * D_);
  float* q_lin = (float*)alloc(sizeof(float) * (size_t)BN_TOK * D_);
  float* kv    = (float*)alloc(sizeof(float) * (size_t)BN_TOK * 2 * D_);
  float* h1    = (float*)alloc(sizeof(float) * (size_t)BN_TOK * MHP_);
  float* h2    = (float*)alloc(sizeof(float) * (size_t)BN_TOK * MHP_);
  float* proj  = (float*)alloc(sizeof(float) * (size_t)BN_TOK * D_);
  // f16 activations
  _Float16* xn_h   = (_Float16*)alloc(sizeof(_Float16) * (size_t)BN_TOK * D_);
  _Float16* src_h  = (_Float16*)alloc(sizeof(_Float16) * (size_t)B_ * M_ * D_);
  _Float16* qh     = (_Float16*)alloc(sizeof(_Float16) * (size_t)B_ * H_ * N_ * HD_);
  _Float16* kh     = (_Float16*)alloc(sizeof(_Float16) * (size_t)B_ * H_ * M_ * HD_);
  _Float16* vt     = (_Float16*)alloc(sizeof(_Float16) * (size_t)B_ * H_ * HD_ * M_);
  _Float16* attn_h = (_Float16*)alloc(sizeof(_Float16) * (size_t)BN_TOK * D_);
  _Float16* hsw_h  = (_Float16*)alloc(sizeof(_Float16) * (size_t)BN_TOK * MHP_);

  auto cast4 = [&](const float* s, _Float16* d, size_t n) {
    size_t n4 = n / 4;
    int blocks = (int)((n4 + 255) / 256); if (blocks > 4096) blocks = 4096;
    cast4_f32_f16<<<blocks, 256, 0, stream>>>((const float4*)s, (v4h*)d, n4);
  };
  auto cast_pad = [&](const float* s, _Float16* d, int sr, int sc, int dr, int dc) {
    size_t total = (size_t)dr * dc;
    int blocks = (int)((total + 255) / 256); if (blocks > 4096) blocks = 4096;
    cast_pad_f32_f16<<<blocks, 256, 0, stream>>>(s, d, sr, sc, dr, dc);
  };
  auto gemm = [&](const _Float16* A, const _Float16* Bw, const float* bias,
                  float* C, int Mr, int Nc, int Kd) {
    dim3 g(Nc / BN, Mr / BM);
    gemm_f16f32<<<g, 256, 0, stream>>>(A, Bw, bias, C, Mr, Nc, Kd);
  };

  // --- weight casts / padding (every call: deterministic) ---
  cast4(sa_qkv_w, w_qkv_h,  (size_t)D_ * 3 * D_);
  cast4(sa_o_w,   w_sao_h,  (size_t)D_ * D_);
  cast4(ca_q_w,   w_caq_h,  (size_t)D_ * D_);
  cast4(ca_kv_w,  w_cakv_h, (size_t)D_ * 2 * D_);
  cast4(ca_o_w,   w_cao_h,  (size_t)D_ * D_);
  cast4(src_tok,  src_h,    (size_t)B_ * M_ * D_);
  cast_pad(mlp_w1, w_m1_h, D_, MH_, D_, MHP_);
  cast_pad(mlp_w2, w_m2_h, D_, MH_, D_, MHP_);
  cast_pad(mlp_w3, w_m3_h, MH_, D_, MHP_, D_);
  pad_bias_f32<<<(MHP_ + 255) / 256, 256, 0, stream>>>(mlp_b1, b1p, MH_, MHP_);
  pad_bias_f32<<<(MHP_ + 255) / 256, 256, 0, stream>>>(mlp_b2, b2p, MH_, MHP_);

  // --- adaLN modulations ---
  mods_kernel<<<dim3(NINE_D / 256, B_), 256, 0, stream>>>(c, ada_w, ada_b, mods);

  const float scale = 0.125f;                     // HD^-0.5
  const int res_blocks = (BN_TOK * D_ / 4 + 255) / 256;

  // ================= self-attention sublayer =================
  rms_mod_cast<<<BN_TOK, 256, 0, stream>>>(x, n1_w, mods, 0 * D_, 1 * D_, xn_h);
  gemm(xn_h, w_qkv_h, sa_qkv_b, qkv, BN_TOK, 3 * D_, D_);
  head_rms_cast<<<8192, 256, 0, stream>>>(qkv, 3 * D_, 0,      sa_qn_w, qh, N_, 0);
  head_rms_cast<<<8192, 256, 0, stream>>>(qkv, 3 * D_, D_,     sa_kn_w, kh, N_, 0);
  head_rms_cast<<<8192, 256, 0, stream>>>(qkv, 3 * D_, 2 * D_, nullptr, vt, N_, 1);
  flash_attn<<<dim3(N_ / 16, H_, B_), 32, 0, stream>>>(qh, kh, vt, attn_h, N_, N_, scale);
  gemm(attn_h, w_sao_h, sa_o_b, proj, BN_TOK, D_, D_);
  residual_gate4<<<res_blocks, 256, 0, stream>>>((const float4*)x, (const float4*)proj,
                                                 mods, 2 * D_, (float4*)x1);

  // ================= cross-attention sublayer =================
  rms_mod_cast<<<BN_TOK, 256, 0, stream>>>(x1, nc_w, mods, 3 * D_, 4 * D_, xn_h);
  gemm(xn_h, w_caq_h, ca_q_b, q_lin, BN_TOK, D_, D_);
  gemm(src_h, w_cakv_h, ca_kv_b, kv, B_ * M_, 2 * D_, D_);
  head_rms_cast<<<8192, 256, 0, stream>>>(q_lin, D_,     0,  ca_qn_w, qh, N_, 0);
  head_rms_cast<<<8192, 256, 0, stream>>>(kv,    2 * D_, 0,  ca_kn_w, kh, M_, 0);
  head_rms_cast<<<8192, 256, 0, stream>>>(kv,    2 * D_, D_, nullptr, vt, M_, 1);
  flash_attn<<<dim3(N_ / 16, H_, B_), 32, 0, stream>>>(qh, kh, vt, attn_h, N_, M_, scale);
  gemm(attn_h, w_cao_h, ca_o_b, proj, BN_TOK, D_, D_);
  residual_gate4<<<res_blocks, 256, 0, stream>>>((const float4*)x1, (const float4*)proj,
                                                 mods, 5 * D_, (float4*)x2);

  // ================= SwiGLU FFN sublayer =================
  rms_mod_cast<<<BN_TOK, 256, 0, stream>>>(x2, n2_w, mods, 6 * D_, 7 * D_, xn_h);
  gemm(xn_h, w_m1_h, b1p, h1, BN_TOK, MHP_, D_);
  gemm(xn_h, w_m2_h, b2p, h2, BN_TOK, MHP_, D_);
  swiglu4<<<4096, 256, 0, stream>>>((const float4*)h1, (const float4*)h2,
                                    (v4h*)hsw_h, (size_t)BN_TOK * MHP_ / 4);
  gemm(hsw_h, w_m3_h, mlp_b3, proj, BN_TOK, D_, MHP_);
  residual_gate4<<<res_blocks, 256, 0, stream>>>((const float4*)x2, (const float4*)proj,
                                                 mods, 8 * D_, (float4*)out);
}
